// ILCMDecoder_62053687492975
// MI455X (gfx1250) — matrix operand order repaired and checked by reading, not compiled, not run on hardware
//
#include <hip/hip_runtime.h>
#include <hip/hip_bf16.h>

typedef __attribute__((ext_vector_type(16))) _Float16 v16h;
typedef __attribute__((ext_vector_type(8)))  _Float16 v8h;
typedef __attribute__((ext_vector_type(8)))  float    v8f;

#define LOG2PI_F 1.8378770664093453f
#define CLAMP_F  1000000.0f

// ---------------------------------------------------------------------------
// Weight transpose + fp32 -> f16 convert:  Wt[n*K + k] = (f16) W[k*N + n]
// ---------------------------------------------------------------------------
__global__ __launch_bounds__(256) void k_transpose_f16(
    const float* __restrict__ W, _Float16* __restrict__ Wt, int K, int N)
{
    size_t idx = (size_t)blockIdx.x * 256 + threadIdx.x;
    if (idx >= (size_t)N * K) return;
    int n = (int)(idx / K);
    int k = (int)(idx % K);
    Wt[idx] = (_Float16)W[(size_t)k * N + n];
}

// ---------------------------------------------------------------------------
// Decoder layer 1: h1[r, j] = relu(e[r,:16] . Wd1[:, j] + bd1[j]),  f16 out
// rows 0..8191 from e1, rows 8192..16383 from e2
// ---------------------------------------------------------------------------
__global__ __launch_bounds__(256) void k_dec_layer1(
    const float* __restrict__ e1, const float* __restrict__ e2,
    const float* __restrict__ Wd1, const float* __restrict__ bd1,
    _Float16* __restrict__ h1)
{
    int r = blockIdx.x;
    const float* e = (r < 8192) ? (e1 + (size_t)r * 16) : (e2 + (size_t)(r - 8192) * 16);
    __shared__ float er[16];
    if (threadIdx.x < 16) er[threadIdx.x] = e[threadIdx.x];
    __syncthreads();
    for (int j = threadIdx.x; j < 1024; j += 256) {
        float acc = bd1[j];
        #pragma unroll
        for (int k = 0; k < 16; ++k) acc += er[k] * Wd1[k * 1024 + j];
        acc = fmaxf(acc, 0.0f);
        h1[(size_t)r * 1024 + j] = (_Float16)acc;
    }
}

// ---------------------------------------------------------------------------
// WMMA GEMM, double-buffered async global->LDS staging (ASYNCcnt path):
//   Out[M,N] = act( A[M,K] * Bt[N,K]^T + bias[N] )
// Block tile 128(M) x 128(N), K-step 32. Per step the block stages
//   A-tile 128x32 f16 (8 KB) and B-tile 128x32 f16 (8 KB) into LDS via
//   global_load_async_to_lds_b128 (16 B per lane), overlapped with WMMA on
//   the other buffer. 8 waves (4Mx2N); each wave computes 32x64 via
//   8x v_wmma_f32_16x16x32_f16 per K-step (2 A frags x 4 B frags).
//
// LDS row stride = 56 halves (112 B): keeps every 16 B async-write chunk
// aligned and spreads b128 row reads across all 64 banks ((r*28)%64 distinct).
//
// Fragment layouts per CDNA5 ISA 7.12.2 (wave32):
//  A 16x32 f16: lane%16 = M row; lanes<16 hold K {0..7,16..23}, lanes>=16 {8..15,24..31}
//  B 32x16 f16: lane%16 = N col; lanes<16 hold K {0..7,16..23}, lanes>=16 {8..15,24..31}
//  C 16x16 f32: lane%16 = N; vgpr v holds M = v + 8*(lane/16)
// ---------------------------------------------------------------------------
#define LDT 56  // halves per LDS tile row (112 bytes)

template<bool RELU, bool F16OUT>
__global__ __launch_bounds__(256) void k_gemm_wmma(
    const _Float16* __restrict__ A,
    const _Float16* __restrict__ Bt,
    const float* __restrict__ bias,
    void* __restrict__ Out,
    int N, int K)
{
    __shared__ _Float16 sA[2][128 * LDT];  // 2 x 14 KB
    __shared__ _Float16 sB[2][128 * LDT];  // 2 x 14 KB

    const int tid   = threadIdx.x;
    const int wave  = tid >> 5;
    const int lane  = tid & 31;
    const int wm    = wave & 3;        // 4 waves along M (32 rows each)
    const int wn    = wave >> 2;       // 2 waves along N (64 cols each)
    const int lhalf = lane >> 4;       // 0 or 1 (K-group half)
    const int l16   = lane & 15;

    const int m0 = blockIdx.y * 128;
    const int n0 = blockIdx.x * 128;

    // --- async-copy work assignment (16-byte chunks) -----------------------
    // A tile: 128 rows x 4 chunks = 512 chunks -> 2 per thread
    // B tile: 128 rows x 4 chunks = 512 chunks -> 2 per thread
    const int cA0 = tid * 2, cA1 = tid * 2 + 1;
    const int rA0 = cA0 >> 2, xA0 = cA0 & 3;
    const int rA1 = cA1 >> 2, xA1 = cA1 & 3;
    const int rB0 = rA0, xB0 = xA0;
    const int rB1 = rA1, xB1 = xA1;

    const _Float16* gA0 = A  + (size_t)(m0 + rA0) * K + xA0 * 8;
    const _Float16* gA1 = A  + (size_t)(m0 + rA1) * K + xA1 * 8;
    const _Float16* gB0 = Bt + (size_t)(n0 + rB0) * K + xB0 * 8;
    const _Float16* gB1 = Bt + (size_t)(n0 + rB1) * K + xB1 * 8;

    // LDS byte offsets (low 32 bits of a generic LDS pointer = LDS offset)
    uint32_t lA0[2], lA1[2], lB0[2], lB1[2];
    #pragma unroll
    for (int s = 0; s < 2; ++s) {
        lA0[s] = (uint32_t)(uintptr_t)&sA[s][rA0 * LDT + xA0 * 8];
        lA1[s] = (uint32_t)(uintptr_t)&sA[s][rA1 * LDT + xA1 * 8];
        lB0[s] = (uint32_t)(uintptr_t)&sB[s][rB0 * LDT + xB0 * 8];
        lB1[s] = (uint32_t)(uintptr_t)&sB[s][rB1 * LDT + xB1 * 8];
    }

    auto issue = [&](int kk, int s) {
        asm volatile("global_load_async_to_lds_b128 %0, %1, off"
                     :: "v"(lA0[s]), "v"(gA0 + kk) : "memory");
        asm volatile("global_load_async_to_lds_b128 %0, %1, off"
                     :: "v"(lA1[s]), "v"(gA1 + kk) : "memory");
        asm volatile("global_load_async_to_lds_b128 %0, %1, off"
                     :: "v"(lB0[s]), "v"(gB0 + kk) : "memory");
        asm volatile("global_load_async_to_lds_b128 %0, %1, off"
                     :: "v"(lB1[s]), "v"(gB1 + kk) : "memory");
    };

    v8f c[2][4] = {};

    issue(0, 0);
    int stage = 0;
    for (int kk = 0; kk < K; kk += 32) {
        asm volatile("s_wait_asynccnt 0x0" ::: "memory");
        __syncthreads();
        if (kk + 32 < K) issue(kk + 32, stage ^ 1);

        const _Float16* pA = &sA[stage][0];
        const _Float16* pB = &sB[stage][0];

        union { v16h v; v8h h[2]; } a[2], b[4];
        #pragma unroll
        for (int ma = 0; ma < 2; ++ma) {
            const _Float16* r = pA + (wm * 32 + ma * 16 + l16) * LDT + lhalf * 8;
            a[ma].h[0] = *(const v8h*)(r);        // K = lhalf*8 + 0..7
            a[ma].h[1] = *(const v8h*)(r + 16);   // K = 16 + lhalf*8 + 0..7
        }
        #pragma unroll
        for (int nb = 0; nb < 4; ++nb) {
            const _Float16* r = pB + (wn * 64 + nb * 16 + l16) * LDT + lhalf * 8;
            b[nb].h[0] = *(const v8h*)(r);        // K = lhalf*8 + 0..7
            b[nb].h[1] = *(const v8h*)(r + 16);   // K = 16 + lhalf*8 + 0..7
        }

        #pragma unroll
        for (int ma = 0; ma < 2; ++ma)
            #pragma unroll
            for (int nb = 0; nb < 4; ++nb)
                c[ma][nb] = __builtin_amdgcn_wmma_f32_16x16x32_f16(
                    false, a[ma].v, false, b[nb].v, (short)0, c[ma][nb], false, false);

        stage ^= 1;
    }

    const int m_w = m0 + wm * 32;
    const int n_w = n0 + wn * 64;
    #pragma unroll
    for (int ma = 0; ma < 2; ++ma) {
        #pragma unroll
        for (int nb = 0; nb < 4; ++nb) {
            const int n  = n_w + nb * 16 + l16;
            const float bn = bias[n];
            #pragma unroll
            for (int v = 0; v < 8; ++v) {
                const int m = m_w + ma * 16 + v + lhalf * 8;
                float val = c[ma][nb][v] + bn;
                if (RELU) val = fmaxf(val, 0.0f);
                if (F16OUT) ((_Float16*)Out)[(size_t)m * N + n] = (_Float16)val;
                else        ((float*)Out)   [(size_t)m * N + n] = val;
            }
        }
    }
}

// ---------------------------------------------------------------------------
// log_p_e1 partial sums: sum of clamp(-0.5 x^2 - 0.5 log(2pi)) over e1
// 128 blocks x 256 threads x 4 elems = 131072 elems
// ---------------------------------------------------------------------------
__global__ __launch_bounds__(256) void k_e1_logp(
    const float* __restrict__ e1, float* __restrict__ partials)
{
    __shared__ float red[256];
    float s = 0.0f;
    #pragma unroll
    for (int t = 0; t < 4; ++t) {
        int idx = blockIdx.x * 1024 + t * 256 + threadIdx.x;
        float x = e1[idx];
        float v = -0.5f * x * x - 0.5f * LOG2PI_F;
        v = fminf(fmaxf(v, -CLAMP_F), CLAMP_F);
        s += v;
    }
    red[threadIdx.x] = s;
    __syncthreads();
    for (int o = 128; o > 0; o >>= 1) {
        if (threadIdx.x < o) red[threadIdx.x] += red[threadIdx.x + o];
        __syncthreads();
    }
    if (threadIdx.x == 0) partials[blockIdx.x] = red[0];
}

// ---------------------------------------------------------------------------
// Flow: per-dimension conditional MLPs -> log_p_e2 partial sums.
// grid = (32, 16): blockIdx.y = i (flow dim), blockIdx.x = batch tile of 256.
// Weights for dim i staged in LDS (all lanes read identical address -> bcast).
// ---------------------------------------------------------------------------
__global__ __launch_bounds__(256) void k_flow(
    const float* __restrict__ e1, const float* __restrict__ e2,
    const float* __restrict__ interv,
    const float* __restrict__ W1, const float* __restrict__ b1,
    const float* __restrict__ W2, const float* __restrict__ b2,
    const float* __restrict__ W3, const float* __restrict__ b3,
    float* __restrict__ partials)
{
    const int i = blockIdx.y;
    const int b = blockIdx.x * 256 + threadIdx.x;

    __shared__ float sW1[16 * 64];
    __shared__ float sW2[64 * 64];
    __shared__ float sW3[64 * 2];
    __shared__ float sb1[64];
    __shared__ float sb2[64];
    __shared__ float sb3[2];
    __shared__ float red[256];

    for (int t = threadIdx.x; t < 16 * 64; t += 256) sW1[t] = W1[i * 1024 + t];
    for (int t = threadIdx.x; t < 64 * 64; t += 256) sW2[t] = W2[i * 4096 + t];
    for (int t = threadIdx.x; t < 128;     t += 256) sW3[t] = W3[i * 128 + t];
    if (threadIdx.x < 64)  sb1[threadIdx.x] = b1[i * 64 + threadIdx.x];
    else if (threadIdx.x < 128) sb2[threadIdx.x - 64] = b2[i * 64 + (threadIdx.x - 64)];
    else if (threadIdx.x < 130) sb3[threadIdx.x - 128] = b3[i * 2 + (threadIdx.x - 128)];
    __syncthreads();

    // ctx = e1 row with component i zeroed
    float cx[16];
    #pragma unroll
    for (int d = 0; d < 16; ++d) {
        float v = e1[(size_t)b * 16 + d];
        cx[d] = (d == i) ? 0.0f : v;
    }

    // h1 = relu(ctx @ W1 + b1)
    float h1r[64];
    #pragma unroll
    for (int hh = 0; hh < 64; ++hh) {
        float acc = sb1[hh];
        #pragma unroll
        for (int d = 0; d < 16; ++d) acc += cx[d] * sW1[d * 64 + hh];
        h1r[hh] = fmaxf(acc, 0.0f);
    }

    // h2 = relu(h1 @ W2 + b2); p = h2 @ W3 + b3 (fused, no h2 array)
    float p0 = sb3[0], p1 = sb3[1];
    for (int k = 0; k < 64; ++k) {
        float acc = sb2[k];
        #pragma unroll
        for (int hh = 0; hh < 64; ++hh) acc += h1r[hh] * sW2[hh * 64 + k];
        acc = fmaxf(acc, 0.0f);
        p0 += acc * sW3[2 * k];
        p1 += acc * sW3[2 * k + 1];
    }

    const float shift = p0;
    const float sp    = fmaxf(p1, 0.0f) + log1pf(expf(-fabsf(p1)));  // softplus
    const float scale = sp + 0.2f;
    float z = (e2[(size_t)b * 16 + i] - shift) / scale;
    z = fminf(fmaxf(z, -CLAMP_F), CLAMP_F);
    const float mask = interv[(size_t)b * 17 + (i + 1)];
    const float val  = (-0.5f * z * z - 0.5f * LOG2PI_F - logf(scale)) * mask;

    red[threadIdx.x] = val;
    __syncthreads();
    for (int o = 128; o > 0; o >>= 1) {
        if (threadIdx.x < o) red[threadIdx.x] += red[threadIdx.x + o];
        __syncthreads();
    }
    if (threadIdx.x == 0) partials[blockIdx.y * 32 + blockIdx.x] = red[0];
}

// ---------------------------------------------------------------------------
// Final deterministic reduction: log_p = sum(flow partials) + sum(e1 partials)
//                                       - log(17) * 8192
// ---------------------------------------------------------------------------
__global__ __launch_bounds__(256) void k_final(
    const float* __restrict__ flowP, const float* __restrict__ e1P,
    float* __restrict__ out_scalar)
{
    __shared__ float red[256];
    float s = 0.0f;
    for (int t = threadIdx.x; t < 512; t += 256) s += flowP[t];
    if (threadIdx.x < 128) s += e1P[threadIdx.x];
    red[threadIdx.x] = s;
    __syncthreads();
    for (int o = 128; o > 0; o >>= 1) {
        if (threadIdx.x < o) red[threadIdx.x] += red[threadIdx.x + o];
        __syncthreads();
    }
    if (threadIdx.x == 0) out_scalar[0] = red[0] - logf(17.0f) * 8192.0f;
}

// ---------------------------------------------------------------------------
extern "C" void kernel_launch(void* const* d_in, const int* in_sizes, int n_in,
                              void* d_out, int out_size, void* d_ws, size_t ws_size,
                              hipStream_t stream)
{
    const float* e1   = (const float*)d_in[0];
    const float* e2   = (const float*)d_in[1];
    const float* itv  = (const float*)d_in[2];
    const float* W1   = (const float*)d_in[3];
    const float* b1   = (const float*)d_in[4];
    const float* W2   = (const float*)d_in[5];
    const float* b2   = (const float*)d_in[6];
    const float* W3   = (const float*)d_in[7];
    const float* b3   = (const float*)d_in[8];
    const float* Wd1  = (const float*)d_in[9];
    const float* bd1  = (const float*)d_in[10];
    const float* Wd2  = (const float*)d_in[11];
    const float* bd2  = (const float*)d_in[12];
    const float* Wd3  = (const float*)d_in[13];
    const float* bd3  = (const float*)d_in[14];
    float* out = (float*)d_out;

    char* ws = (char*)d_ws;
    _Float16* Wd2t  = (_Float16*)(ws);                          //  2 MB (1024x1024)
    _Float16* Wd3t  = (_Float16*)(ws + ((size_t)2  << 20));     //  6 MB (3072x1024)
    _Float16* h1    = (_Float16*)(ws + ((size_t)8  << 20));     // 32 MB (16384x1024)
    _Float16* h2    = (_Float16*)(ws + ((size_t)40 << 20));     // 32 MB (16384x1024)
    float*    flowP = (float*)   (ws + ((size_t)72 << 20));     // 512 floats
    float*    e1P   = flowP + 512;                              // 128 floats

    // Weight transposes (fp32 -> f16, N-major for contiguous WMMA B fragments)
    k_transpose_f16<<<(1024 * 1024 + 255) / 256, 256, 0, stream>>>(Wd2, Wd2t, 1024, 1024);
    k_transpose_f16<<<(3072 * 1024 + 255) / 256, 256, 0, stream>>>(Wd3, Wd3t, 1024, 3072);

    // Decoder layer 1 (K=16, FMA) for e1||e2 stacked as 16384 rows
    k_dec_layer1<<<16384, 256, 0, stream>>>(e1, e2, Wd1, bd1, h1);

    // Decoder layer 2: h2 = relu(h1 @ Wd2 + bd2), f16 out  (M=16384,N=1024,K=1024)
    k_gemm_wmma<true, true><<<dim3(1024 / 128, 16384 / 128), 256, 0, stream>>>(
        h1, Wd2t, bd2, (void*)h2, 1024, 1024);

    // Decoder layer 3: out = h2 @ Wd3 + bd3, fp32 straight to d_out
    // (rows 0..8191 -> x1_hat, rows 8192..16383 -> x2_hat; layout is contiguous)
    k_gemm_wmma<false, false><<<dim3(3072 / 128, 16384 / 128), 256, 0, stream>>>(
        h2, Wd3t, bd3, (void*)out, 3072, 1024);

    // log-prob terms
    k_e1_logp<<<128, 256, 0, stream>>>(e1, e1P);
    k_flow<<<dim3(32, 16), 256, 0, stream>>>(e1, e2, itv, W1, b1, W2, b2, W3, b3, flowP);
    k_final<<<1, 256, 0, stream>>>(flowP, e1P, out + (size_t)2 * 8192 * 3072);

    (void)in_sizes; (void)n_in; (void)out_size; (void)ws_size;
}
